// Mamba_41102837022812
// MI455X (gfx1250) — matrix operand order repaired and checked by reading, not compiled, not run on hardware
//
#include <hip/hip_runtime.h>
#include <hip/hip_bf16.h>

// ---------------------------------------------------------------------------
// Mamba block for MI455X (gfx1250, wave32, WMMA).
// GEMMs in f16 WMMA (f32 accumulate); selective scan as a chunked two-pass
// linear-recurrence scan (parallelism B*C*NCHUNK instead of B*C).
// ---------------------------------------------------------------------------

typedef __attribute__((ext_vector_type(16))) _Float16 v16h;
typedef __attribute__((ext_vector_type(8)))  _Float16 v8h;
typedef __attribute__((ext_vector_type(8)))  float    v8f;

#define C_DIM 256
#define N_STATE 16
#define R_EXT 4
#define B_SZ 2
#define SW (C_DIM + 2 * N_STATE)   // 288
#define CLEN 128                   // scan chunk length (divides L)
#define PF_AHEAD 16                // prefetch distance (rows)

// ---------------- fragment loaders (layouts per CDNA5 ISA) ------

// A-matrix 16x32 f16, row-major source, leading dim lda.
// lane 0-15: row=lane,    halves 0..7 = K[0..7],  halves 8..15 = K[16..23]
// lane16-31: row=lane-16, halves 0..7 = K[8..15], halves 8..15 = K[24..31]
__device__ __forceinline__ v16h load_a_frag(const _Float16* __restrict__ A, int lda,
                                            int mTile, int k0, int lane) {
  int row = mTile + (lane & 15);
  int kb  = k0 + ((lane >> 4) << 3);
  const _Float16* p = A + (size_t)row * lda + kb;
  v16h a;
  ((v8h*)&a)[0] = *(const v8h*)(p);        // K kb..kb+7
  ((v8h*)&a)[1] = *(const v8h*)(p + 16);   // K kb+16..kb+23
  return a;
}

// B-matrix 32x16 f16 from transposed weights Wt[N,K], leading dim ldb(=K).
// lane 0-15: col=lane,    halves 0..7 = K[0..7],  halves 8..15 = K[16..23]
// lane16-31: col=lane-16, halves 0..7 = K[8..15], halves 8..15 = K[24..31]
__device__ __forceinline__ v16h load_b_frag(const _Float16* __restrict__ Wt, int ldb,
                                            int nTile, int k0, int lane) {
  int col = nTile + (lane & 15);
  int kb  = k0 + ((lane >> 4) << 3);
  const _Float16* p = Wt + (size_t)col * ldb + kb;
  v16h b;
  ((v8h*)&b)[0] = *(const v8h*)(p);
  ((v8h*)&b)[1] = *(const v8h*)(p + 16);
  return b;
}

// One wave computes a 32x32 output tile: 2x2 grid of 16x16 WMMA accumulators.
__device__ __forceinline__ void wave_gemm_32x32(const _Float16* __restrict__ A, int lda,
                                                const _Float16* __restrict__ Bt, int ldb,
                                                int K, int mTile, int nTile, int lane,
                                                v8f acc[2][2]) {
  for (int k0 = 0; k0 < K; k0 += 32) {
    v16h a0 = load_a_frag(A, lda, mTile,      k0, lane);
    v16h a1 = load_a_frag(A, lda, mTile + 16, k0, lane);
    v16h b0 = load_b_frag(Bt, ldb, nTile,      k0, lane);
    v16h b1 = load_b_frag(Bt, ldb, nTile + 16, k0, lane);
    acc[0][0] = __builtin_amdgcn_wmma_f32_16x16x32_f16(false, a0, false, b0,
                                                       (short)0, acc[0][0], false, false);
    acc[0][1] = __builtin_amdgcn_wmma_f32_16x16x32_f16(false, a0, false, b1,
                                                       (short)0, acc[0][1], false, false);
    acc[1][0] = __builtin_amdgcn_wmma_f32_16x16x32_f16(false, a1, false, b0,
                                                       (short)0, acc[1][0], false, false);
    acc[1][1] = __builtin_amdgcn_wmma_f32_16x16x32_f16(false, a1, false, b1,
                                                       (short)0, acc[1][1], false, false);
  }
}

// ---------------- block-wide (256-thread) reduction ------------------------

__device__ __forceinline__ float block_sum256(float v, float* sm) {
  __syncthreads();
  #pragma unroll
  for (int o = 16; o > 0; o >>= 1) v += __shfl_down(v, o, 32);
  if ((threadIdx.x & 31) == 0) sm[threadIdx.x >> 5] = v;
  __syncthreads();
  float s = sm[0];
  #pragma unroll
  for (int i = 1; i < 8; ++i) s += sm[i];
  return s;
}

__device__ __forceinline__ float softplus_f(float x) {
  return (x > 20.f) ? x : log1pf(__expf(x));
}

// ---------------- kernels --------------------------------------------------

// Weight convert+transpose: W[K,N] f32 -> Wt[N,K] f16.
__global__ void k_wt(const float* __restrict__ W, _Float16* __restrict__ Wt, int K, int N) {
  int idx = blockIdx.x * blockDim.x + threadIdx.x;
  if (idx >= K * N) return;
  int n = idx / K;
  int k = idx - n * K;
  Wt[idx] = (_Float16)W[(size_t)k * N + n];
}

// LayerNorm (row of 256) -> f16.
__global__ void k_ln_f16(const float* __restrict__ x, const float* __restrict__ scale,
                         const float* __restrict__ bias, _Float16* __restrict__ out) {
  __shared__ float sm[8];
  size_t row = blockIdx.x;
  int t = threadIdx.x;
  float v = x[row * C_DIM + t];
  float mean = block_sum256(v, sm) * (1.f / C_DIM);
  float d = v - mean;
  float var = block_sum256(d * d, sm) * (1.f / C_DIM);
  float r = rsqrtf(var + 1e-5f);
  out[row * C_DIM + t] = (_Float16)(d * r * scale[t] + bias[t]);
}

// GEMM1: proj = silu(xn @ W_in + b_in); split into xg (f16 + f32) and z (f32).
__global__ void k_gemm1(const _Float16* __restrict__ A, const _Float16* __restrict__ Bt,
                        const float* __restrict__ bias, _Float16* __restrict__ xg16,
                        float* __restrict__ xg32, float* __restrict__ z32) {
  const int K = C_DIM, Ncols = 2 * C_DIM;
  int gw = (blockIdx.x * blockDim.x + threadIdx.x) >> 5;
  int lane = threadIdx.x & 31;
  int nt = Ncols / 32;
  int mTile = (gw / nt) * 32, nTile = (gw % nt) * 32;
  v8f acc[2][2] = {};
  wave_gemm_32x32(A, K, Bt, K, K, mTile, nTile, lane, acc);
  #pragma unroll
  for (int i = 0; i < 2; ++i)
    #pragma unroll
    for (int j = 0; j < 2; ++j) {
      int n  = nTile + j * 16 + (lane & 15);
      int mb = mTile + i * 16 + ((lane >> 4) << 3);
      float bn = bias[n];
      #pragma unroll
      for (int r = 0; r < 8; ++r) {
        float v = acc[i][j][r] + bn;
        float s = v / (1.f + __expf(-v));      // silu
        size_t m = (size_t)(mb + r);
        if (n < C_DIM) {
          xg16[m * C_DIM + n] = (_Float16)s;
          xg32[m * C_DIM + n] = s;
        } else {
          z32[m * C_DIM + (n - C_DIM)] = s;
        }
      }
    }
}

// GEMM2: ssm = xg @ W_ssm + b_ssm  (f32 out, width 288).
__global__ void k_gemm2(const _Float16* __restrict__ A, const _Float16* __restrict__ Bt,
                        const float* __restrict__ bias, float* __restrict__ ssm) {
  const int K = C_DIM, Ncols = SW;
  int gw = (blockIdx.x * blockDim.x + threadIdx.x) >> 5;
  int lane = threadIdx.x & 31;
  int nt = Ncols / 32;
  int mTile = (gw / nt) * 32, nTile = (gw % nt) * 32;
  v8f acc[2][2] = {};
  wave_gemm_32x32(A, K, Bt, K, K, mTile, nTile, lane, acc);
  #pragma unroll
  for (int i = 0; i < 2; ++i)
    #pragma unroll
    for (int j = 0; j < 2; ++j) {
      int n  = nTile + j * 16 + (lane & 15);
      int mb = mTile + i * 16 + ((lane >> 4) << 3);
      float bn = bias[n];
      #pragma unroll
      for (int r = 0; r < 8; ++r)
        ssm[(size_t)(mb + r) * Ncols + n] = acc[i][j][r] + bn;
    }
}

// ---- chunked selective scan -----------------------------------------------
// Recurrence per (b,c,n):  h_t = dA_t * h_{t-1} + dt_t*u_t*B_t[n]
// Total time = R_EXT * L steps; the data stream repeats every L rows.
// Chunk k covers steps [k*CLEN, (k+1)*CLEN); CLEN divides L so a chunk never
// crosses a replica boundary (replica r = k*CLEN / L).
// Chunk-state arrays indexed: [((b*NCH + k)*N_STATE + n)*C_DIM + c]

// Pass 1: per-chunk local scan from h=0; record end state and decay product.
__global__ void k_scan_part(const float* __restrict__ ssm, const float* __restrict__ xg32,
                            const float* __restrict__ dt_bias_in, const float* __restrict__ As_log,
                            float* __restrict__ hloc, float* __restrict__ Pch, int L) {
  int k = blockIdx.x, b = blockIdx.y, c = threadIdx.x;
  int NCH = gridDim.x;
  int cpr = L / CLEN;                       // chunks per replica
  int l0  = (k % cpr) * CLEN;
  float Ac[N_STATE], h[N_STATE], P[N_STATE];
  #pragma unroll
  for (int n = 0; n < N_STATE; ++n) {
    Ac[n] = -__expf(As_log[c * N_STATE + n]);
    h[n] = 0.f;
    P[n] = 1.f;
  }
  float dtb = dt_bias_in[c];
  size_t row0 = (size_t)b * L + l0;
  for (int s = 0; s < CLEN; ++s) {
    size_t row = row0 + s;
    const float* srow = ssm + row * SW;
    if (s + PF_AHEAD < CLEN) {
      __builtin_prefetch(srow + PF_AHEAD * SW, 0, 1);
      __builtin_prefetch(xg32 + (row + PF_AHEAD) * C_DIM, 0, 1);
    }
    float dt = softplus_f(srow[c] + dtb);
    float du = dt * xg32[row * C_DIM + c];
    const float* Br = srow + C_DIM;
    #pragma unroll
    for (int n = 0; n < N_STATE; ++n) {
      float dA = __expf(dt * Ac[n]);
      h[n] = dA * h[n] + du * Br[n];
      P[n] *= dA;
    }
  }
  size_t base = (size_t)(b * NCH + k) * N_STATE * C_DIM + c;
  #pragma unroll
  for (int n = 0; n < N_STATE; ++n) {
    hloc[base + (size_t)n * C_DIM] = h[n];
    Pch [base + (size_t)n * C_DIM] = P[n];
  }
}

// Pass 2: sequential carry across chunks (cheap): record true incoming state.
__global__ void k_scan_carry(const float* __restrict__ hloc, const float* __restrict__ Pch,
                             float* __restrict__ hin, int NCH) {
  int b = blockIdx.x, c = threadIdx.x;
  float h[N_STATE];
  #pragma unroll
  for (int n = 0; n < N_STATE; ++n) h[n] = 0.f;
  for (int k = 0; k < NCH; ++k) {
    size_t base = (size_t)(b * NCH + k) * N_STATE * C_DIM + c;
    #pragma unroll
    for (int n = 0; n < N_STATE; ++n) {
      size_t idx = base + (size_t)n * C_DIM;
      hin[idx] = h[n];
      h[n] = Pch[idx] * h[n] + hloc[idx];
    }
  }
}

// Pass 3: re-run each chunk from its true h_in, emit gamma-scaled y per replica.
// ypart layout: [r][b][l][c]
__global__ void k_scan_fix(const float* __restrict__ ssm, const float* __restrict__ xg32,
                           const float* __restrict__ dt_bias_in, const float* __restrict__ As_log,
                           const float* __restrict__ gamma, const float* __restrict__ hin,
                           float* __restrict__ ypart, int L) {
  int k = blockIdx.x, b = blockIdx.y, c = threadIdx.x;
  int NCH = gridDim.x, B = gridDim.y;
  int cpr = L / CLEN;
  int r   = k / cpr;
  int l0  = (k % cpr) * CLEN;
  float g = gamma[c * R_EXT + r];
  float Ac[N_STATE], h[N_STATE];
  size_t base = (size_t)(b * NCH + k) * N_STATE * C_DIM + c;
  #pragma unroll
  for (int n = 0; n < N_STATE; ++n) {
    Ac[n] = -__expf(As_log[c * N_STATE + n]);
    h[n] = hin[base + (size_t)n * C_DIM];
  }
  float dtb = dt_bias_in[c];
  size_t row0 = (size_t)b * L + l0;
  float* yrow0 = ypart + (((size_t)r * B + b) * L + l0) * C_DIM + c;
  for (int s = 0; s < CLEN; ++s) {
    size_t row = row0 + s;
    const float* srow = ssm + row * SW;
    if (s + PF_AHEAD < CLEN) {
      __builtin_prefetch(srow + PF_AHEAD * SW, 0, 1);
      __builtin_prefetch(xg32 + (row + PF_AHEAD) * C_DIM, 0, 1);
    }
    float dt = softplus_f(srow[c] + dtb);
    float du = dt * xg32[row * C_DIM + c];
    const float* Br = srow + C_DIM;
    const float* Cr = srow + C_DIM + N_STATE;
    float y = 0.f;
    #pragma unroll
    for (int n = 0; n < N_STATE; ++n) {
      float dA = __expf(dt * Ac[n]);
      h[n] = dA * h[n] + du * Br[n];
      y += h[n] * Cr[n];
    }
    yrow0[(size_t)s * C_DIM] = g * y;
  }
}

// Combine replicas + D*u term, LayerNorm, gate with z -> f16 (input to GEMM3).
__global__ void k_ln_gate(const float* __restrict__ ypart, const float* __restrict__ xg32,
                          const float* __restrict__ z, const float* __restrict__ gamma,
                          const float* __restrict__ Ds,
                          const float* __restrict__ scale, const float* __restrict__ bias,
                          _Float16* __restrict__ out, int L, int B) {
  __shared__ float sm[8];
  size_t row = blockIdx.x;               // token index m = b*L + l
  int t = threadIdx.x;
  int b = (int)(row / L), l = (int)(row % L);
  float gsum = 0.f;
  #pragma unroll
  for (int r = 0; r < R_EXT; ++r) gsum += gamma[t * R_EXT + r];
  float v = gsum * Ds[t] * xg32[row * C_DIM + t];
  #pragma unroll
  for (int r = 0; r < R_EXT; ++r)
    v += ypart[(((size_t)r * B + b) * L + l) * C_DIM + t];
  float mean = block_sum256(v, sm) * (1.f / C_DIM);
  float d = v - mean;
  float var = block_sum256(d * d, sm) * (1.f / C_DIM);
  float ln = d * rsqrtf(var + 1e-5f) * scale[t] + bias[t];
  out[row * C_DIM + t] = (_Float16)(ln * z[row * C_DIM + t]);
}

// GEMM3: out = shortcut + gated @ W_out + b_out  (f32 out).
__global__ void k_gemm3(const _Float16* __restrict__ A, const _Float16* __restrict__ Bt,
                        const float* __restrict__ bias, const float* __restrict__ shortcut,
                        float* __restrict__ out) {
  const int K = C_DIM, Ncols = C_DIM;
  int gw = (blockIdx.x * blockDim.x + threadIdx.x) >> 5;
  int lane = threadIdx.x & 31;
  int nt = Ncols / 32;
  int mTile = (gw / nt) * 32, nTile = (gw % nt) * 32;
  v8f acc[2][2] = {};
  wave_gemm_32x32(A, K, Bt, K, K, mTile, nTile, lane, acc);
  #pragma unroll
  for (int i = 0; i < 2; ++i)
    #pragma unroll
    for (int j = 0; j < 2; ++j) {
      int n  = nTile + j * 16 + (lane & 15);
      int mb = mTile + i * 16 + ((lane >> 4) << 3);
      float bn = bias[n];
      #pragma unroll
      for (int r = 0; r < 8; ++r) {
        size_t m = (size_t)(mb + r);
        out[m * Ncols + n] = shortcut[m * Ncols + n] + acc[i][j][r] + bn;
      }
    }
}

// ---------------- host launcher --------------------------------------------

extern "C" void kernel_launch(void* const* d_in, const int* in_sizes, int n_in,
                              void* d_out, int out_size, void* d_ws, size_t ws_size,
                              hipStream_t stream) {
  const float* x            = (const float*)d_in[0];
  // d_in[1] = batch_size (device int scalar) — B_SZ is a compile-time constant.
  const float* ln_in_scale  = (const float*)d_in[2];
  const float* ln_in_bias   = (const float*)d_in[3];
  const float* W_in         = (const float*)d_in[4];
  const float* b_in         = (const float*)d_in[5];
  const float* W_ssm        = (const float*)d_in[6];
  const float* b_ssm        = (const float*)d_in[7];
  const float* ssm_pbias    = (const float*)d_in[8];
  const float* As_log       = (const float*)d_in[9];
  const float* Ds           = (const float*)d_in[10];
  const float* gamma        = (const float*)d_in[11];
  const float* ln_out_scale = (const float*)d_in[12];
  const float* ln_out_bias  = (const float*)d_in[13];
  const float* W_out        = (const float*)d_in[14];
  const float* b_out        = (const float*)d_in[15];
  float* out = (float*)d_out;

  const int M   = in_sizes[0] / C_DIM;     // 8192 tokens
  const int L   = M / B_SZ;                // 4096
  const int NCH = (R_EXT * L) / CLEN;      // 128 chunks per (b,c) stream

  // workspace carve-up (256B aligned)
  char* w = (char*)d_ws;
  auto nxt = [&](size_t bytes) -> char* {
    char* p = w;
    w += (bytes + 255) & ~(size_t)255;
    return p;
  };
  _Float16* xn16    = (_Float16*)nxt((size_t)M * C_DIM * 2);
  _Float16* wt_in   = (_Float16*)nxt((size_t)(2 * C_DIM) * C_DIM * 2);
  _Float16* wt_ssm  = (_Float16*)nxt((size_t)SW * C_DIM * 2);
  _Float16* wt_out  = (_Float16*)nxt((size_t)C_DIM * C_DIM * 2);
  _Float16* xg16    = (_Float16*)nxt((size_t)M * C_DIM * 2);
  float*    xg32    = (float*)nxt((size_t)M * C_DIM * 4);
  float*    z32     = (float*)nxt((size_t)M * C_DIM * 4);
  float*    ssm32   = (float*)nxt((size_t)M * SW * 4);
  float*    ypart   = (float*)nxt((size_t)R_EXT * M * C_DIM * 4);
  float*    hloc    = (float*)nxt((size_t)B_SZ * NCH * N_STATE * C_DIM * 4);
  float*    Pch     = (float*)nxt((size_t)B_SZ * NCH * N_STATE * C_DIM * 4);
  float*    hin     = (float*)nxt((size_t)B_SZ * NCH * N_STATE * C_DIM * 4);
  _Float16* gated16 = (_Float16*)nxt((size_t)M * C_DIM * 2);

  // 1. weight transpose/convert to f16 [N,K]
  k_wt<<<(2 * C_DIM * C_DIM) / 256, 256, 0, stream>>>(W_in, wt_in, C_DIM, 2 * C_DIM);
  k_wt<<<(SW * C_DIM) / 256, 256, 0, stream>>>(W_ssm, wt_ssm, C_DIM, SW);
  k_wt<<<(C_DIM * C_DIM) / 256, 256, 0, stream>>>(W_out, wt_out, C_DIM, C_DIM);

  // 2. LN(x) -> f16
  k_ln_f16<<<M, 256, 0, stream>>>(x, ln_in_scale, ln_in_bias, xn16);

  // 3. GEMM1 + silu + split
  {
    int waves = (M / 32) * ((2 * C_DIM) / 32);
    k_gemm1<<<waves / 8, 256, 0, stream>>>(xn16, wt_in, b_in, xg16, xg32, z32);
  }

  // 4. GEMM2
  {
    int waves = (M / 32) * (SW / 32);
    k_gemm2<<<waves / 8, 256, 0, stream>>>(xg16, wt_ssm, b_ssm, ssm32);
  }

  // 5. chunked selective scan: local scans -> carry -> fixup with true states
  {
    dim3 grid(NCH, B_SZ);
    k_scan_part<<<grid, C_DIM, 0, stream>>>(ssm32, xg32, ssm_pbias, As_log, hloc, Pch, L);
    k_scan_carry<<<B_SZ, C_DIM, 0, stream>>>(hloc, Pch, hin, NCH);
    k_scan_fix<<<grid, C_DIM, 0, stream>>>(ssm32, xg32, ssm_pbias, As_log, gamma, hin,
                                           ypart, L);
  }

  // 6. combine replicas + D*u, LN(y), gate with z -> f16
  k_ln_gate<<<M, 256, 0, stream>>>(ypart, xg32, z32, gamma, Ds,
                                   ln_out_scale, ln_out_bias, gated16, L, B_SZ);

  // 7. GEMM3 + shortcut
  {
    int waves = (M / 32) * (C_DIM / 32);
    k_gemm3<<<waves / 8, 256, 0, stream>>>(gated16, wt_out, b_out, x, out);
  }
}
